// CQAttention_61873298866894
// MI455X (gfx1250) — matrix operand order repaired and checked by reading, not compiled, not run on hardware
//
#include <hip/hip_runtime.h>

// ---------------------------------------------------------------------------
// CQAttention for MI455X (gfx1250): all GEMMs on V_WMMA_F32_16X16X32_F16.
// f16 operands staged in workspace with layouts matching the WMMA fragment
// striping (ISA 7.12.2), f32 accumulation, f32 softmax. Final GEMM stages its
// shared B operand (out_w^T) into LDS with GLOBAL_LOAD_ASYNC_TO_LDS_B128.
// ---------------------------------------------------------------------------

#define NB   16
#define LC   1024
#define LQ   512
#define DIM  512
#define K4D  2048
#define NEG_INF (-1e30f)

typedef _Float16 half16v __attribute__((ext_vector_type(16)));
typedef _Float16 half8v  __attribute__((ext_vector_type(8)));
typedef float    float8v __attribute__((ext_vector_type(8)));

// ---- WMMA fragment helpers (wave32) ---------------------------------------
// A: 16x32 f16, row-major source. lane<16: rows M=lane, K={0..7,16..23};
// lane>=16: same rows, K={8..15,24..31}. Two 16B loads per lane.
__device__ __forceinline__ half16v wmma_ld_a(const _Float16* __restrict__ A,
                                             int lda, int row0, int k0) {
    int lane = threadIdx.x & 31;
    int m  = row0 + (lane & 15);
    int kb = k0 + ((lane >> 4) << 3);
    const _Float16* p = A + (size_t)m * lda + kb;
    half8v lo = *(const half8v*)(p);
    half8v hi = *(const half8v*)(p + 16);
    return __builtin_shufflevector(lo, hi, 0, 1, 2, 3, 4, 5, 6, 7,
                                   8, 9, 10, 11, 12, 13, 14, 15);
}

// B: 32x16 f16, K-major source (ldb = N-stride). Lane L holds column N=L%16:
// lane<16: K={0..7,16..23}; lane>=16: K={8..15,24..31} (strided gather).
__device__ __forceinline__ half16v wmma_ld_b(const _Float16* __restrict__ B,
                                             int ldb, int k0, int col0) {
    int lane = threadIdx.x & 31;
    int n  = col0 + (lane & 15);
    int kb = k0 + ((lane >> 4) << 3);
    half16v v;
    #pragma unroll
    for (int s = 0; s < 8; ++s) {
        v[s]     = B[(size_t)(kb + s) * ldb + n];
        v[s + 8] = B[(size_t)(kb + 16 + s) * ldb + n];
    }
    return v;
}

__device__ __forceinline__ float8v wmma16(half16v a, half16v b, float8v c) {
    return __builtin_amdgcn_wmma_f32_16x16x32_f16(false, a, false, b,
                                                  (short)0, c, false, false);
}

// ---- precompute / conversion kernels --------------------------------------
__global__ __launch_bounds__(256) void k_convert_C(
    const float* __restrict__ C, const float* __restrict__ w3,
    _Float16* __restrict__ Ch, _Float16* __restrict__ Cw3h,
    _Float16* __restrict__ Cath) {
    size_t idx = (size_t)blockIdx.x * 256 + threadIdx.x;  // over NB*LC*DIM
    float v = C[idx];
    int k = (int)(idx & (DIM - 1));
    size_t row = idx >> 9;            // b*LC + i
    Ch[idx]   = (_Float16)v;
    Cw3h[idx] = (_Float16)(v * w3[k]);
    Cath[row * K4D + k] = (_Float16)v;   // concat segment 0 = C
}

__global__ __launch_bounds__(256) void k_convert_Q(
    const float* __restrict__ Q, _Float16* __restrict__ Qh,
    _Float16* __restrict__ QTh) {
    size_t idx = (size_t)blockIdx.x * 256 + threadIdx.x;  // over NB*LQ*DIM
    float v = Q[idx];
    int k = (int)(idx & (DIM - 1));
    size_t rj = idx >> 9;             // b*LQ + j
    int j = (int)(rj & (LQ - 1));
    int b = (int)(rj >> 9);
    Qh[idx] = (_Float16)v;
    QTh[((size_t)b * DIM + k) * LQ + j] = (_Float16)v;   // K-major Q^T
}

__global__ __launch_bounds__(256) void k_convert_W(
    const float* __restrict__ out_w, _Float16* __restrict__ Wt) {
    size_t idx = (size_t)blockIdx.x * 256 + threadIdx.x;  // over DIM*K4D
    int n  = (int)(idx / K4D);
    int kk = (int)(idx - (size_t)n * K4D);
    Wt[(size_t)kk * DIM + n] = (_Float16)out_w[idx];      // out_w^T, K-major
}

// One wave per row: dot(X[row,:], w[:DIM])
__global__ __launch_bounds__(256) void k_rowdot(
    const float* __restrict__ X, const float* __restrict__ w,
    float* __restrict__ out) {
    int row  = blockIdx.x * 8 + (threadIdx.x >> 5);
    int lane = threadIdx.x & 31;
    const float* x = X + (size_t)row * DIM;
    float s = 0.f;
    #pragma unroll 4
    for (int k = lane; k < DIM; k += 32) s += x[k] * w[k];
    #pragma unroll
    for (int o = 16; o; o >>= 1) s += __shfl_xor(s, o, 32);
    if (lane == 0) out[row] = s;
}

// ---- GEMM 1: S = (C*w3) Q^T + rowC[i] + rowQ[j] ---------------------------
__global__ __launch_bounds__(256) void k_gemm_S(
    const _Float16* __restrict__ Cw3h, const _Float16* __restrict__ QTh,
    const float* __restrict__ rowC, const float* __restrict__ rowQ,
    float* __restrict__ S) {
    int b = blockIdx.z;
    int wave = threadIdx.x >> 5, lane = threadIdx.x & 31;
    int row0 = blockIdx.y * 64  + (wave & 1) * 32;
    int col0 = blockIdx.x * 128 + (wave >> 1) * 32;
    const _Float16* A  = Cw3h + (size_t)b * LC * DIM;
    const _Float16* Bm = QTh  + (size_t)b * DIM * LQ;
    float8v acc[2][2] = {};
    for (int k = 0; k < DIM; k += 32) {
        half16v a0 = wmma_ld_a(A, DIM, row0,      k);
        half16v a1 = wmma_ld_a(A, DIM, row0 + 16, k);
        half16v b0 = wmma_ld_b(Bm, LQ, k, col0);
        half16v b1 = wmma_ld_b(Bm, LQ, k, col0 + 16);
        acc[0][0] = wmma16(a0, b0, acc[0][0]);
        acc[0][1] = wmma16(a0, b1, acc[0][1]);
        acc[1][0] = wmma16(a1, b0, acc[1][0]);
        acc[1][1] = wmma16(a1, b1, acc[1][1]);
    }
    float* Sb = S + (size_t)b * LC * LQ;
    int n_l = lane & 15, m_l = (lane >> 4) * 8;
    #pragma unroll
    for (int mi = 0; mi < 2; ++mi)
        #pragma unroll
        for (int ni = 0; ni < 2; ++ni) {
            int n = col0 + ni * 16 + n_l;
            float rq = rowQ[b * LQ + n];
            #pragma unroll
            for (int r = 0; r < 8; ++r) {
                int m = row0 + mi * 16 + m_l + r;
                Sb[(size_t)m * LQ + n] = acc[mi][ni][r] + rowC[b * LC + m] + rq;
            }
        }
}

// ---- softmax helpers ------------------------------------------------------
__device__ __forceinline__ float block_reduce(float v, float* sm, bool do_max) {
    #pragma unroll
    for (int o = 16; o; o >>= 1) {
        float t = __shfl_xor(v, o, 32);
        v = do_max ? fmaxf(v, t) : (v + t);
    }
    __syncthreads();
    if ((threadIdx.x & 31) == 0) sm[threadIdx.x >> 5] = v;
    __syncthreads();
    float r = sm[0];
    #pragma unroll
    for (int i = 1; i < 8; ++i) r = do_max ? fmaxf(r, sm[i]) : (r + sm[i]);
    return r;
}

// softmax over j (axis=2) with qmask -> S1h f16 (B,LC,LQ)
__global__ __launch_bounds__(256) void k_softmax_rows(
    const float* __restrict__ S, const float* __restrict__ qmask,
    _Float16* __restrict__ S1h) {
    __shared__ float sm[8];
    int row = blockIdx.x;               // b*LC + i
    int b = row >> 10;
    const float* Srow = S + (size_t)row * LQ;
    const float* qm = qmask + (size_t)b * LQ;
    int t = threadIdx.x;
    float m0 = qm[t], m1 = qm[t + 256];
    float l0 = Srow[t] * m0 + (1.f - m0) * NEG_INF;
    float l1 = Srow[t + 256] * m1 + (1.f - m1) * NEG_INF;
    float mx = block_reduce(fmaxf(l0, l1), sm, true);
    float e0 = __expf(l0 - mx), e1 = __expf(l1 - mx);
    float inv = 1.f / block_reduce(e0 + e1, sm, false);
    _Float16* D = S1h + (size_t)row * LQ;
    D[t]       = (_Float16)(e0 * inv);
    D[t + 256] = (_Float16)(e1 * inv);
}

// softmax over i (axis=1) with cmask -> S2Th f16 transposed (B,LQ,LC)
__global__ __launch_bounds__(256) void k_softmax_cols(
    const float* __restrict__ S, const float* __restrict__ cmask,
    _Float16* __restrict__ S2Th) {
    __shared__ float sm[8];
    int col = blockIdx.x;               // b*LQ + j
    int b = col >> 9, j = col & (LQ - 1);
    const float* Sb = S + (size_t)b * LC * LQ + j;
    const float* cm = cmask + (size_t)b * LC;
    int t = threadIdx.x;
    float l[4];
    #pragma unroll
    for (int kk = 0; kk < 4; ++kk) {
        int i = t + kk * 256;
        float m = cm[i];
        l[kk] = Sb[(size_t)i * LQ] * m + (1.f - m) * NEG_INF;
    }
    float mx = fmaxf(fmaxf(l[0], l[1]), fmaxf(l[2], l[3]));
    mx = block_reduce(mx, sm, true);
    float e[4], s = 0.f;
    #pragma unroll
    for (int kk = 0; kk < 4; ++kk) { e[kk] = __expf(l[kk] - mx); s += e[kk]; }
    float inv = 1.f / block_reduce(s, sm, false);
    _Float16* D = S2Th + (size_t)col * LC;
    #pragma unroll
    for (int kk = 0; kk < 4; ++kk) D[t + kk * 256] = (_Float16)(e[kk] * inv);
}

// ---- GEMM 2: S2TC = S2^T C -> f16 (B,LQ,DIM) ------------------------------
__global__ __launch_bounds__(256) void k_gemm_s2tc(
    const _Float16* __restrict__ S2Th, const _Float16* __restrict__ Ch,
    _Float16* __restrict__ S2TCh) {
    int b = blockIdx.z;
    int wave = threadIdx.x >> 5, lane = threadIdx.x & 31;
    int row0 = blockIdx.y * 64  + (wave & 1) * 32;
    int col0 = blockIdx.x * 128 + (wave >> 1) * 32;
    const _Float16* A  = S2Th + (size_t)b * LQ * LC;
    const _Float16* Bm = Ch   + (size_t)b * LC * DIM;
    float8v acc[2][2] = {};
    for (int k = 0; k < LC; k += 32) {
        half16v a0 = wmma_ld_a(A, LC, row0,      k);
        half16v a1 = wmma_ld_a(A, LC, row0 + 16, k);
        half16v b0 = wmma_ld_b(Bm, DIM, k, col0);
        half16v b1 = wmma_ld_b(Bm, DIM, k, col0 + 16);
        acc[0][0] = wmma16(a0, b0, acc[0][0]);
        acc[0][1] = wmma16(a0, b1, acc[0][1]);
        acc[1][0] = wmma16(a1, b0, acc[1][0]);
        acc[1][1] = wmma16(a1, b1, acc[1][1]);
    }
    _Float16* Ob = S2TCh + (size_t)b * LQ * DIM;
    int n_l = lane & 15, m_l = (lane >> 4) * 8;
    #pragma unroll
    for (int mi = 0; mi < 2; ++mi)
        #pragma unroll
        for (int ni = 0; ni < 2; ++ni) {
            int n = col0 + ni * 16 + n_l;
            #pragma unroll
            for (int r = 0; r < 8; ++r) {
                int m = row0 + mi * 16 + m_l + r;
                Ob[(size_t)m * DIM + n] = (_Float16)acc[mi][ni][r];
            }
        }
}

// ---- GEMM 3+4 fused: A_att = S1 Q, Bm = S1 S2TC; build concat f16 ---------
__global__ __launch_bounds__(256) void k_gemm_AB(
    const _Float16* __restrict__ S1h, const _Float16* __restrict__ Qh,
    const _Float16* __restrict__ S2TCh, const float* __restrict__ C,
    _Float16* __restrict__ Cath) {
    int b = blockIdx.z;
    int wave = threadIdx.x >> 5, lane = threadIdx.x & 31;
    int row0 = blockIdx.y * 64  + (wave & 1) * 32;
    int col0 = blockIdx.x * 128 + (wave >> 1) * 32;
    const _Float16* A  = S1h   + (size_t)b * LC * LQ;
    const _Float16* Bq = Qh    + (size_t)b * LQ * DIM;
    const _Float16* Bs = S2TCh + (size_t)b * LQ * DIM;
    float8v accA[2][2] = {}, accB[2][2] = {};
    for (int k = 0; k < LQ; k += 32) {
        half16v a0 = wmma_ld_a(A, LQ, row0,      k);
        half16v a1 = wmma_ld_a(A, LQ, row0 + 16, k);
        half16v q0 = wmma_ld_b(Bq, DIM, k, col0);
        half16v q1 = wmma_ld_b(Bq, DIM, k, col0 + 16);
        half16v s0 = wmma_ld_b(Bs, DIM, k, col0);
        half16v s1 = wmma_ld_b(Bs, DIM, k, col0 + 16);
        accA[0][0] = wmma16(a0, q0, accA[0][0]);
        accA[0][1] = wmma16(a0, q1, accA[0][1]);
        accA[1][0] = wmma16(a1, q0, accA[1][0]);
        accA[1][1] = wmma16(a1, q1, accA[1][1]);
        accB[0][0] = wmma16(a0, s0, accB[0][0]);
        accB[0][1] = wmma16(a0, s1, accB[0][1]);
        accB[1][0] = wmma16(a1, s0, accB[1][0]);
        accB[1][1] = wmma16(a1, s1, accB[1][1]);
    }
    const float* Cb = C + (size_t)b * LC * DIM;
    int n_l = lane & 15, m_l = (lane >> 4) * 8;
    #pragma unroll
    for (int mi = 0; mi < 2; ++mi)
        #pragma unroll
        for (int ni = 0; ni < 2; ++ni) {
            int n = col0 + ni * 16 + n_l;
            #pragma unroll
            for (int r = 0; r < 8; ++r) {
                int m = row0 + mi * 16 + m_l + r;
                float cv = Cb[(size_t)m * DIM + n];
                float av = accA[mi][ni][r];
                float bv = accB[mi][ni][r];
                _Float16* cr = Cath + ((size_t)b * LC + m) * K4D;
                cr[DIM + n]     = (_Float16)av;        // A
                cr[2 * DIM + n] = (_Float16)(cv * av); // C*A
                cr[3 * DIM + n] = (_Float16)(cv * bv); // C*Bm
            }
        }
}

// ---- GEMM 5: out = [C|A|C*A|C*Bm] @ out_w^T + out_b -----------------------
// B operand (Wt) is shared by all 256 M-tiles: stage each 128K x 128N chunk
// into LDS with async global->LDS loads (ASYNCcnt), read B frags via DS.
__global__ __launch_bounds__(256) void k_gemm_out(
    const _Float16* __restrict__ Cath, const _Float16* __restrict__ Wt,
    const float* __restrict__ bias, float* __restrict__ Out) {
    __shared__ _Float16 sB[128 * 128];           // 32 KB of the 320 KB WGP LDS
    int wave = threadIdx.x >> 5, lane = threadIdx.x & 31;
    int row0   = blockIdx.y * 64 + (wave & 1) * 32;   // global row b*LC+i
    int colblk = blockIdx.x * 128;
    int colloc = (wave >> 1) * 32;                    // wave's N offset in LDS
    unsigned int lds_base = (unsigned int)(unsigned long long)(uintptr_t)sB;
    float8v acc[2][2] = {};
    for (int kc = 0; kc < K4D; kc += 128) {
        // async stage Wt[kc:kc+128, colblk:colblk+128] -> sB (each thread 8x16B)
        #pragma unroll
        for (int t = 0; t < 8; ++t) {
            int cidx = threadIdx.x + 256 * t;         // 2048 16B chunks
            int r  = cidx >> 4;                       // row in chunk (K)
            int cc = (cidx & 15) << 3;                // col in halves
            const _Float16* g = Wt + (size_t)(kc + r) * DIM + colblk + cc;
            unsigned int la = lds_base + (unsigned int)(((r << 7) + cc) << 1);
            asm volatile("global_load_async_to_lds_b128 %0, %1, off"
                         :: "v"(la), "v"((unsigned long long)(uintptr_t)g)
                         : "memory");
        }
        asm volatile("s_wait_asynccnt 0x0" ::: "memory");
        __syncthreads();
        #pragma unroll
        for (int kk = 0; kk < 128; kk += 32) {
            int k = kc + kk;
            __builtin_prefetch(Cath + (size_t)(row0 + lane) * K4D + k + 128, 0, 1);
            half16v a0 = wmma_ld_a(Cath, K4D, row0,      k);
            half16v a1 = wmma_ld_a(Cath, K4D, row0 + 16, k);
            half16v b0 = wmma_ld_b(sB, 128, kk, colloc);
            half16v b1 = wmma_ld_b(sB, 128, kk, colloc + 16);
            acc[0][0] = wmma16(a0, b0, acc[0][0]);
            acc[0][1] = wmma16(a0, b1, acc[0][1]);
            acc[1][0] = wmma16(a1, b0, acc[1][0]);
            acc[1][1] = wmma16(a1, b1, acc[1][1]);
        }
        __syncthreads();
    }
    int n_l = lane & 15, m_l = (lane >> 4) * 8;
    #pragma unroll
    for (int mi = 0; mi < 2; ++mi)
        #pragma unroll
        for (int ni = 0; ni < 2; ++ni) {
            int n = colblk + colloc + ni * 16 + n_l;
            float bb = bias[n];
            #pragma unroll
            for (int r = 0; r < 8; ++r) {
                int m = row0 + mi * 16 + m_l + r;
                Out[(size_t)m * DIM + n] = acc[mi][ni][r] + bb;
            }
        }
}

// ---------------------------------------------------------------------------
extern "C" void kernel_launch(void* const* d_in, const int* in_sizes, int n_in,
                              void* d_out, int out_size, void* d_ws, size_t ws_size,
                              hipStream_t stream) {
    const float* C     = (const float*)d_in[0];
    const float* Q     = (const float*)d_in[1];
    const float* cmask = (const float*)d_in[2];
    const float* qmask = (const float*)d_in[3];
    const float* w     = (const float*)d_in[4];
    const float* out_w = (const float*)d_in[5];
    const float* out_b = (const float*)d_in[6];
    float* out = (float*)d_out;

    // workspace layout (~196 MB)
    char* p = (char*)d_ws;
    _Float16* Ch    = (_Float16*)p; p += (size_t)NB * LC * DIM * 2;
    _Float16* Cw3h  = (_Float16*)p; p += (size_t)NB * LC * DIM * 2;
    _Float16* Qh    = (_Float16*)p; p += (size_t)NB * LQ * DIM * 2;
    _Float16* QTh   = (_Float16*)p; p += (size_t)NB * DIM * LQ * 2;
    _Float16* Wt    = (_Float16*)p; p += (size_t)K4D * DIM * 2;
    float*    rowC  = (float*)p;    p += (size_t)NB * LC * 4;
    float*    rowQ  = (float*)p;    p += (size_t)NB * LQ * 4;
    float*    S     = (float*)p;    p += (size_t)NB * LC * LQ * 4;
    _Float16* S1h   = (_Float16*)p; p += (size_t)NB * LC * LQ * 2;
    _Float16* S2Th  = (_Float16*)p; p += (size_t)NB * LQ * LC * 2;
    _Float16* S2TCh = (_Float16*)p; p += (size_t)NB * LQ * DIM * 2;
    _Float16* Cath  = (_Float16*)p; p += (size_t)NB * LC * K4D * 2;

    // 1) conversions / transposes
    k_convert_C<<<(NB * LC * DIM) / 256, 256, 0, stream>>>(C, w + 2 * DIM, Ch, Cw3h, Cath);
    k_convert_Q<<<(NB * LQ * DIM) / 256, 256, 0, stream>>>(Q, Qh, QTh);
    k_convert_W<<<(DIM * K4D) / 256, 256, 0, stream>>>(out_w, Wt);
    // 2) rank-1 bias terms
    k_rowdot<<<(NB * LC) / 8, 256, 0, stream>>>(C, w, rowC);
    k_rowdot<<<(NB * LQ) / 8, 256, 0, stream>>>(Q, w + DIM, rowQ);
    // 3) similarity matrix
    k_gemm_S<<<dim3(LQ / 128, LC / 64, NB), 256, 0, stream>>>(Cw3h, QTh, rowC, rowQ, S);
    // 4) both softmaxes
    k_softmax_rows<<<NB * LC, 256, 0, stream>>>(S, qmask, S1h);
    k_softmax_cols<<<NB * LQ, 256, 0, stream>>>(S, cmask, S2Th);
    // 5) S2^T C
    k_gemm_s2tc<<<dim3(DIM / 128, LQ / 64, NB), 256, 0, stream>>>(S2Th, Ch, S2TCh);
    // 6) A and Bm fused (shared S1 fragments), build concat
    k_gemm_AB<<<dim3(DIM / 128, LC / 64, NB), 256, 0, stream>>>(S1h, Qh, S2TCh, C, Cath);
    // 7) final projection + bias (async-LDS staged B operand)
    k_gemm_out<<<dim3(DIM / 128, (NB * LC) / 64, 1), 256, 0, stream>>>(Cath, Wt, out_b, out);
}
